// ResMoELoRALinear_48627619725935
// MI455X (gfx1250) — matrix operand order repaired and checked by reading, not compiled, not run on hardware
//
#include <hip/hip_runtime.h>

// Problem constants (fixed shapes from the reference)
#define BATCH 4
#define SEQ   2048
#define IN_F  4096
#define OUT_F 4096
#define RES   64
#define N_EXP 16

#define M_TOK (BATCH * SEQ)          // 8192 tokens
#define KC    (IN_F + N_EXP * RES)   // 5120 concatenated-K

typedef __bf16 bf16x8  __attribute__((ext_vector_type(8)));
typedef __bf16 bf16x16 __attribute__((ext_vector_type(16)));
typedef float  f32x4   __attribute__((ext_vector_type(4)));
typedef float  f32x8   __attribute__((ext_vector_type(8)));
typedef unsigned int u32x2 __attribute__((ext_vector_type(2)));
typedef unsigned int u32x4 __attribute__((ext_vector_type(4)));
typedef int v4i __attribute__((vector_size(16)));   // matches builtin proto

// gfx1250 async global->LDS path (ASYNCcnt-tracked), with safe fallback.
#if defined(__HIP_DEVICE_COMPILE__) && \
    __has_builtin(__builtin_amdgcn_global_load_async_to_lds_b128)
#define USE_ASYNC_LDS 1
#else
#define USE_ASYNC_LDS 0
#endif

#if USE_ASYNC_LDS
using gv4i = __attribute__((address_space(1))) v4i;   // global int4
using lv4i = __attribute__((address_space(3))) v4i;   // LDS int4
#endif

__device__ __forceinline__ unsigned short f2bf(float f) {
  unsigned int u = __builtin_bit_cast(unsigned int, f);
  u += 0x7FFFu + ((u >> 16) & 1u);   // round-to-nearest-even
  return (unsigned short)(u >> 16);
}

__device__ __forceinline__ void wait_async0() {
#if defined(__HIP_DEVICE_COMPILE__) && \
    __has_builtin(__builtin_amdgcn_s_wait_asynccnt)
  __builtin_amdgcn_s_wait_asynccnt(0);
#elif defined(__HIP_DEVICE_COMPILE__)
  asm volatile("s_wait_asynccnt 0" ::: "memory");
#endif
}

// One lane-wise 16-byte async copy: global -> LDS, tracked by ASYNCcnt.
// Flat LDS addresses keep the LDS offset in addr[31:0] (ISA aperture rules),
// so uintptr_t -> AS(3) inttoptr yields the correct LDS address.
__device__ __forceinline__ void async_copy16(const unsigned short* g,
                                             unsigned short* l) {
#if USE_ASYNC_LDS
  gv4i* gp = reinterpret_cast<gv4i*>(reinterpret_cast<uintptr_t>(g));
  lv4i* lp = reinterpret_cast<lv4i*>(reinterpret_cast<uintptr_t>(l));
  __builtin_amdgcn_global_load_async_to_lds_b128(gp, lp, 0, 0);
#else
  (void)g; (void)l;
#endif
}

// ---------------------------------------------------------------------------
// Kernel 1: W_base [OUT_F, IN_F] f32 -> Wcat[d][0:IN_F] bf16
// ---------------------------------------------------------------------------
__global__ __launch_bounds__(256) void prep_w_kernel(
    const float* __restrict__ Wb, unsigned short* __restrict__ Wcat) {
  int gid  = blockIdx.x * 256 + threadIdx.x;
  long base = (long)gid * 8;                       // 8 elems / thread
  int d = (int)(base >> 12);                       // / IN_F
  int i = (int)(base & (IN_F - 1));
  f32x4 a = *(const f32x4*)(Wb + base);
  f32x4 b = *(const f32x4*)(Wb + base + 4);
  u32x4 p;
  p.x = (unsigned)f2bf(a.x) | ((unsigned)f2bf(a.y) << 16);
  p.y = (unsigned)f2bf(a.z) | ((unsigned)f2bf(a.w) << 16);
  p.z = (unsigned)f2bf(b.x) | ((unsigned)f2bf(b.y) << 16);
  p.w = (unsigned)f2bf(b.z) | ((unsigned)f2bf(b.w) << 16);
  *(u32x4*)(Wcat + (long)d * KC + i) = p;
}

// ---------------------------------------------------------------------------
// Kernel 2: B [N_EXP, OUT_F, RES] f32 -> Wcat[d][IN_F + n*RES + r] bf16
// ---------------------------------------------------------------------------
__global__ __launch_bounds__(256) void prep_b_kernel(
    const float* __restrict__ Bw, unsigned short* __restrict__ Wcat) {
  int gid  = blockIdx.x * 256 + threadIdx.x;
  long base = (long)gid * 4;                       // 4 elems / thread
  int n   = (int)(base >> 18);                     // / (OUT_F*RES)
  int rem = (int)(base & (OUT_F * RES - 1));
  int d   = rem >> 6;                              // / RES
  int r   = rem & (RES - 1);
  f32x4 v = *(const f32x4*)(Bw + base);
  u32x2 p;
  p.x = (unsigned)f2bf(v.x) | ((unsigned)f2bf(v.y) << 16);
  p.y = (unsigned)f2bf(v.z) | ((unsigned)f2bf(v.w) << 16);
  *(u32x2*)(Wcat + (long)d * KC + IN_F + n * RES + r) = p;
}

// ---------------------------------------------------------------------------
// Kernel 3: per-token router + reservoir + Xcat build.
//   Xcat[t][0:IN_F]           = bf16(x[t])
//   Xcat[t][IN_F + n*RES + r] = bf16(routing[t][n] * (x[t] . A[r]))
// One 256-thread block (8 wave32) per token.
// ---------------------------------------------------------------------------
__global__ __launch_bounds__(256) void router_kernel(
    const float* __restrict__ x, const float* __restrict__ A,
    const float* __restrict__ Wr, unsigned short* __restrict__ Xcat) {
  __shared__ float xs[IN_F];
  __shared__ float red[N_EXP + RES];   // 0..15 router logits, 16..79 res_hidden
  __shared__ float routing_s[N_EXP];

  const int t   = blockIdx.x;
  const int tid = threadIdx.x;
  const float* xrow = x + (long)t * IN_F;
  unsigned short* xc = Xcat + (long)t * KC;

  // Phase 1: stage x in LDS (f32) and emit bf16 copy into Xcat.
#pragma unroll
  for (int i = 0; i < 4; ++i) {
    int c = (tid + i * 256) * 4;
    f32x4 v = *(const f32x4*)(xrow + c);
    *(f32x4*)(xs + c) = v;
    u32x2 p;
    p.x = (unsigned)f2bf(v.x) | ((unsigned)f2bf(v.y) << 16);
    p.y = (unsigned)f2bf(v.z) | ((unsigned)f2bf(v.w) << 16);
    *(u32x2*)(xc + c) = p;
  }
  __syncthreads();

  // Phase 2: 80 dot products (16 router rows + 64 reservoir rows), wave32 each.
  const int lane = tid & 31, w = tid >> 5;
#pragma unroll
  for (int j = 0; j < 10; ++j) {
    int row = w * 10 + j;
    const float* wrow = (row < N_EXP) ? (Wr + (long)row * IN_F)
                                      : (A + (long)(row - N_EXP) * IN_F);
    float s = 0.f;
    for (int i = lane; i < IN_F; i += 32) s = fmaf(xs[i], wrow[i], s);
    s += __shfl_xor(s, 16); s += __shfl_xor(s, 8); s += __shfl_xor(s, 4);
    s += __shfl_xor(s, 2);  s += __shfl_xor(s, 1);
    if (lane == 0) red[row] = s;
  }
  __syncthreads();

  // Phase 3: softmax + top-2 + renormalize (single lane, 16 experts).
  if (tid == 0) {
    float p[N_EXP];
    float mx = red[0];
    for (int n = 1; n < N_EXP; ++n) mx = fmaxf(mx, red[n]);
    float sum = 0.f;
    for (int n = 0; n < N_EXP; ++n) { p[n] = __expf(red[n] - mx); sum += p[n]; }
    float inv = 1.f / sum;
    for (int n = 0; n < N_EXP; ++n) p[n] *= inv;
    int t1 = 0;
    for (int n = 1; n < N_EXP; ++n) if (p[n] > p[t1]) t1 = n;
    int t2 = (t1 == 0) ? 1 : 0;
    for (int n = 0; n < N_EXP; ++n) if (n != t1 && p[n] > p[t2]) t2 = n;
    float denom = p[t1] + p[t2] + 1e-6f;
    for (int n = 0; n < N_EXP; ++n)
      routing_s[n] = (n == t1 || n == t2) ? p[n] / denom : 0.f;
  }
  __syncthreads();

  // Phase 4: Xcat tail = routing[n] * res_hidden[r].
#pragma unroll
  for (int i = 0; i < 4; ++i) {
    int j = tid + i * 256;                  // 0..1023
    int n = j >> 6, r = j & (RES - 1);
    xc[IN_F + j] = f2bf(routing_s[n] * red[N_EXP + r]);
  }
}

// ---------------------------------------------------------------------------
// Kernel 4: fused bf16 WMMA GEMM with bias.
//   out[m][d] = bias[d] + sum_k Xcat[m][k] * Wcat[d][k]
// Block tile 128x128, K-step 64, 8 wave32 (2x4 wave grid), each wave 64x32.
// Async path: double-buffered LDS, GLOBAL_LOAD_ASYNC_TO_LDS_B128 prefetch of
// tile i+1 overlapped with WMMA on tile i; 1 barrier + 1 s_wait_asynccnt/iter.
// ---------------------------------------------------------------------------
#define LDK 72   // padded K stride in LDS (bank-conflict avoidance), 16B rows

__device__ __forceinline__ void compute_tile(
    const unsigned short* __restrict__ Xsb,
    const unsigned short* __restrict__ Wsb,
    int wm, int wn, int half, int l16, f32x8 acc[4][2]) {
#pragma unroll
  for (int ks = 0; ks < 64; ks += 32) {
    // A fragments (16x32 bf16, ISA 7.12.2): e<8 -> k=ks+8*half+e,
    // e>=8 -> k=ks+16+8*half+(e-8): two 16B LDS chunks (+0, +32B).
    bf16x16 av[4];
#pragma unroll
    for (int mt = 0; mt < 4; ++mt) {
      int row = wm * 64 + mt * 16 + l16;
      const bf16x8* p = (const bf16x8*)(Xsb + row * LDK + ks + half * 8);
      bf16x8 lo = p[0], hi = p[2];
      av[mt] = __builtin_shufflevector(lo, hi, 0, 1, 2, 3, 4, 5, 6, 7,
                                       8, 9, 10, 11, 12, 13, 14, 15);
    }
#pragma unroll
    for (int nt = 0; nt < 2; ++nt) {
      // B fragment (32x16): lane n = l16, same k order as A (+0, +32B chunks).
      int row = wn * 32 + nt * 16 + l16;
      const bf16x8* p = (const bf16x8*)(Wsb + row * LDK + ks + half * 8);
      bf16x8 lo = p[0], hi = p[2];
      bf16x16 bv = __builtin_shufflevector(lo, hi, 0, 1, 2, 3, 4, 5, 6, 7,
                                           8, 9, 10, 11, 12, 13, 14, 15);
#pragma unroll
      for (int mt = 0; mt < 4; ++mt) {
        acc[mt][nt] = __builtin_amdgcn_wmma_f32_16x16x32_bf16(
            false, av[mt], false, bv, (short)0, acc[mt][nt], false, false);
      }
    }
  }
}

__global__ __launch_bounds__(256) void moe_gemm_kernel(
    const unsigned short* __restrict__ Xcat,
    const unsigned short* __restrict__ Wcat,
    const float* __restrict__ bias,
    float* __restrict__ out) {
  __shared__ __align__(16) unsigned short Xs[2][128 * LDK];
  __shared__ __align__(16) unsigned short Ws[2][128 * LDK];

  const int tid  = threadIdx.x;
  const int lane = tid & 31, w = tid >> 5;
  const int wm = w & 1, wn = w >> 1;          // 2 waves on M, 4 on N
  const int half = lane >> 4, l16 = lane & 15;
  const int m0 = blockIdx.y * 128;
  const int n0 = blockIdx.x * 128;

  f32x8 acc[4][2] = {};                        // 4 M-tiles x 2 N-tiles

#if USE_ASYNC_LDS
  // ---- async double-buffered pipeline ----
  // Per tile: 1024 16B chunks for X and W each; 4 chunks/thread/tile.
#define ISSUE_TILE(K0, BUF)                                                   \
  do {                                                                        \
    _Pragma("unroll")                                                         \
    for (int i = 0; i < 4; ++i) {                                             \
      int c = tid + i * 256;                                                  \
      int row = c >> 3;                                                       \
      int col = (c & 7) * 8;                                                  \
      async_copy16(Xcat + (long)(m0 + row) * KC + (K0) + col,                 \
                   &Xs[(BUF)][row * LDK + col]);                              \
      async_copy16(Wcat + (long)(n0 + row) * KC + (K0) + col,                 \
                   &Ws[(BUF)][row * LDK + col]);                              \
    }                                                                         \
  } while (0)

  ISSUE_TILE(0, 0);
  for (int it = 0; it < KC / 64; ++it) {
    const int k0  = it * 64;
    const int buf = it & 1;
    wait_async0();       // tile `buf` fully written (this wave's portion)
    __syncthreads();     // all waves: writes visible, prior reads of buf^1 done
    if (k0 + 64 < KC) ISSUE_TILE(k0 + 64, buf ^ 1);  // overlap with compute
    compute_tile(Xs[buf], Ws[buf], wm, wn, half, l16, acc);
  }
#undef ISSUE_TILE
#else
  // ---- fallback: global -> VGPR -> ds_store staging ----
  for (int k0 = 0; k0 < KC; k0 += 64) {
    u32x4 xg[4], wg[4];
#pragma unroll
    for (int i = 0; i < 4; ++i) {
      int c = tid + i * 256;
      int row = c >> 3;
      int col = (c & 7) * 8;
      xg[i] = *(const u32x4*)(Xcat + (long)(m0 + row) * KC + k0 + col);
      wg[i] = *(const u32x4*)(Wcat + (long)(n0 + row) * KC + k0 + col);
    }
    __syncthreads();
#pragma unroll
    for (int i = 0; i < 4; ++i) {
      int c = tid + i * 256;
      int row = c >> 3;
      int col = (c & 7) * 8;
      *(u32x4*)(&Xs[0][row * LDK + col]) = xg[i];
      *(u32x4*)(&Ws[0][row * LDK + col]) = wg[i];
    }
    __syncthreads();
    if (k0 + 64 < KC) {
      __builtin_prefetch(Xcat + (long)(m0 + (tid >> 1)) * KC + k0 + 64, 0, 0);
      __builtin_prefetch(Wcat + (long)(n0 + (tid >> 1)) * KC + k0 + 64, 0, 0);
    }
    compute_tile(Xs[0], Ws[0], wm, wn, half, l16, acc);
  }
#endif

  // Epilogue: C/D layout -> VGPR r holds M = r + 8*half, N = l16.
#pragma unroll
  for (int nt = 0; nt < 2; ++nt) {
    int col = n0 + wn * 32 + nt * 16 + l16;
    float bv = bias[col];
#pragma unroll
    for (int mt = 0; mt < 4; ++mt) {
      int mbase = m0 + wm * 64 + mt * 16 + half * 8;
#pragma unroll
      for (int r = 0; r < 8; ++r) {
        out[(long)(mbase + r) * OUT_F + col] = acc[mt][nt][r] + bv;
      }
    }
  }
}

// ---------------------------------------------------------------------------
// Host launch. Workspace layout (needs 120 MB):
//   Wcat bf16 [OUT_F][KC]  : 4096*5120*2  = 40 MB at offset 0
//   Xcat bf16 [M_TOK][KC]  : 8192*5120*2  = 80 MB after it
// ---------------------------------------------------------------------------
extern "C" void kernel_launch(void* const* d_in, const int* in_sizes, int n_in,
                              void* d_out, int out_size, void* d_ws, size_t ws_size,
                              hipStream_t stream) {
  const float* x  = (const float*)d_in[0];
  const float* Wb = (const float*)d_in[1];
  const float* bb = (const float*)d_in[2];
  const float* A  = (const float*)d_in[3];
  const float* Bw = (const float*)d_in[4];
  const float* Wr = (const float*)d_in[5];
  float* out = (float*)d_out;

  unsigned short* Wcat = (unsigned short*)d_ws;
  unsigned short* Xcat = Wcat + (size_t)OUT_F * KC;

  prep_w_kernel<<<(OUT_F * IN_F / 8) / 256, 256, 0, stream>>>(Wb, Wcat);
  prep_b_kernel<<<(N_EXP * OUT_F * RES / 4) / 256, 256, 0, stream>>>(Bw, Wcat);
  router_kernel<<<M_TOK, 256, 0, stream>>>(x, A, Wr, Xcat);
  moe_gemm_kernel<<<dim3(OUT_F / 128, M_TOK / 128), 256, 0, stream>>>(
      Xcat, Wcat, bb, out);
}
